// Critic_MATD3_Attention_Potential_20624432956264
// MI455X (gfx1250) — matrix operand order repaired and checked by reading, not compiled, not run on hardware
//
#include <hip/hip_runtime.h>
#include <hip/hip_bf16.h>

typedef __attribute__((ext_vector_type(16))) _Float16 v16h;
typedef __attribute__((ext_vector_type(8)))  _Float16 half8;
typedef __attribute__((ext_vector_type(8)))  float    v8f;
typedef __attribute__((ext_vector_type(4)))  unsigned int u32x4;
typedef __attribute__((ext_vector_type(8)))  int i32x8;
typedef __attribute__((ext_vector_type(4)))  int i32x4;

#define BB   2048
#define HH   256
#define DD   64
#define KHEADS 4
#define INW  576
#define SENC 512
#define TDM_SLAB 128

#if __has_builtin(__builtin_amdgcn_tensor_load_to_lds)
#define HAVE_TDM 1
#else
#define HAVE_TDM 0
#endif

// ---------------- WMMA helpers (gfx1250 wave32 layouts) ----------------
__device__ __forceinline__ v8f wmma16(v16h a, v16h b, v8f c) {
    return __builtin_amdgcn_wmma_f32_16x16x32_f16(false, a, false, b, (short)0, c, false, false);
}

// A fragment: row-major (M x K) f16, element i -> K = kbase + i + 16*lh
__device__ __forceinline__ v16h ldfragA(const _Float16* __restrict__ base, int ld,
                                        int row, int kbase, int lq, int lh) {
    const _Float16* p = base + (size_t)(row + lq) * ld + kbase + 16 * lh;
    half8 lo = *(const half8*)p;
    half8 hi = *(const half8*)(p + 8);
    v16h r;
#pragma unroll
    for (int i = 0; i < 8; i++) { r[i] = lo[i]; r[i + 8] = hi[i]; }
    return r;
}

// B fragment from transposed weights: row-major (N x K) f16, element i -> K = kbase + i + 16*lh
__device__ __forceinline__ v16h ldfragB(const _Float16* __restrict__ base, int ld,
                                        int col, int kbase, int lq, int lh) {
    const _Float16* p = base + (size_t)(col + lq) * ld + kbase + 16 * lh;
    half8 lo = *(const half8*)p;
    half8 hi = *(const half8*)(p + 8);
    v16h r;
#pragma unroll
    for (int i = 0; i < 8; i++) { r[i] = lo[i]; r[i + 8] = hi[i]; }
    return r;
}

// ---------------- pack kernels ----------------
__global__ void __launch_bounds__(256)
pack_sa_kernel(const float* __restrict__ s, const float* __restrict__ a,
               _Float16* __restrict__ sa) {
    int idx = blockIdx.x * 256 + threadIdx.x;
    if (idx >= BB * INW) return;
    int b = idx / INW, c = idx - b * INW;
    float v;
    if (c < SENC) v = s[((size_t)(c >> 6) * BB + b) * 64 + (c & 63)];
    else { int cc = c - SENC; v = a[((size_t)(cc >> 3) * BB + b) * 8 + (cc & 7)]; }
    sa[idx] = (_Float16)v;
}

__global__ void __launch_bounds__(256)
wt_pack_kernel(const float* __restrict__ W, _Float16* __restrict__ Wt,
               int K, int N, int total) {
    int idx = blockIdx.x * 256 + threadIdx.x;
    if (idx >= total) return;
    int kn = K * N;
    int bt = idx / kn, r = idx - bt * kn;
    int k = r / N, n = r - k * N;
    Wt[(size_t)bt * kn + (size_t)n * K + k] = (_Float16)W[idx];
}

// ---------------- Fw softmax over batch per agent ----------------
__global__ void __launch_bounds__(256)
fw_kernel(const float* __restrict__ s, float* __restrict__ Fw) {
    int n = blockIdx.x, tid = threadIdx.x;
    __shared__ float red[256];
    __shared__ float bmax_s, bsum_s;
    float loc[8]; float lmax = -1e30f;
#pragma unroll
    for (int j = 0; j < 8; j++) {
        int b = tid + j * 256;
        const float* row = s + ((size_t)n * BB + b) * 64;
        float fx = row[44], fy = row[45], fz = row[46];
        float tx = row[12], ty = row[13], tz = row[14];
        float n1 = sqrtf(fx * fx + fy * fy + fz * fz);
        float n2 = sqrtf(tx * tx + ty * ty + tz * tz);
        float dt = fx * tx + fy * ty + fz * tz;
        float sim = dt / (fmaxf(n1, 1e-8f) * fmaxf(n2, 1e-8f));
        loc[j] = sim * sqrtf(n2);
        lmax = fmaxf(lmax, loc[j]);
    }
    red[tid] = lmax; __syncthreads();
    for (int st = 128; st > 0; st >>= 1) { if (tid < st) red[tid] = fmaxf(red[tid], red[tid + st]); __syncthreads(); }
    if (tid == 0) bmax_s = red[0];
    __syncthreads();
    float bmax = bmax_s, lsum = 0.f;
    __syncthreads();
#pragma unroll
    for (int j = 0; j < 8; j++) { loc[j] = __expf(loc[j] - bmax); lsum += loc[j]; }
    red[tid] = lsum; __syncthreads();
    for (int st = 128; st > 0; st >>= 1) { if (tid < st) red[tid] += red[tid + st]; __syncthreads(); }
    if (tid == 0) bsum_s = red[0];
    __syncthreads();
    float inv = 1.f / bsum_s;
#pragma unroll
    for (int j = 0; j < 8; j++) Fw[(size_t)n * BB + tid + j * 256] = loc[j] * inv;
}

__global__ void __launch_bounds__(256)
fwsum_kernel(const float* __restrict__ Fw, float* __restrict__ FwSum) {
    int b = blockIdx.x * 256 + threadIdx.x;
    if (b >= BB) return;
    float sum = 0.f;
#pragma unroll
    for (int n = 0; n < 8; n++) sum += Fw[(size_t)n * BB + b];
    FwSum[b] = sum;
}

__global__ void __launch_bounds__(256)
vsum_kernel(const _Float16* __restrict__ valT, float* __restrict__ Vsum) {
    int kd = blockIdx.x, tid = threadIdx.x;
    __shared__ float red[256];
    const _Float16* row = valT + (size_t)kd * BB;
    float sum = 0.f;
    for (int c = tid; c < BB; c += 256) sum += (float)row[c];
    red[tid] = sum; __syncthreads();
    for (int st = 128; st > 0; st >>= 1) { if (tid < st) red[tid] += red[tid + st]; __syncthreads(); }
    if (tid == 0) Vsum[kd] = red[0];
}

// ---------------- generic WMMA GEMM (32x32 wave tile) ----------------
// ACT: 0 none, 1 relu, 2 leaky_relu(0.01). TRANS: store out transposed (N x M).
template <int ACT, bool TRANS>
__global__ void __launch_bounds__(256)
gemm_f16_kernel(const _Float16* __restrict__ A, int lda,
                const _Float16* __restrict__ W, int ldb,
                const float* __restrict__ bias,
                _Float16* __restrict__ out, int ldo,
                int M, int N, int K,
                long wStride, long biasStride, long outStride) {
    int batch = blockIdx.y;
    W += (size_t)batch * wStride;
    out += (size_t)batch * outStride;
    const float* bptr = bias + (size_t)batch * biasStride;

    int wave = threadIdx.x >> 5, lane = threadIdx.x & 31;
    int tilesM = M >> 5;
    int t = blockIdx.x * (blockDim.x >> 5) + wave;
    if (t >= tilesM * (N >> 5)) return;
    int tM = t % tilesM, tN = t / tilesM;
    int m0 = tM << 5, n0 = tN << 5;
    int lq = lane & 15, lh = lane >> 4;

    v8f acc[2][2];
#pragma unroll
    for (int i = 0; i < 2; i++)
#pragma unroll
        for (int j = 0; j < 2; j++) acc[i][j] = (v8f){0.f,0.f,0.f,0.f,0.f,0.f,0.f,0.f};

    for (int k = 0; k < K; k += 32) {
        v16h a0 = ldfragA(A, lda, m0, k, lq, lh);
        v16h a1 = ldfragA(A, lda, m0 + 16, k, lq, lh);
        v16h b0 = ldfragB(W, ldb, n0, k, lq, lh);
        v16h b1 = ldfragB(W, ldb, n0 + 16, k, lq, lh);
        acc[0][0] = wmma16(a0, b0, acc[0][0]);
        acc[0][1] = wmma16(a0, b1, acc[0][1]);
        acc[1][0] = wmma16(a1, b0, acc[1][0]);
        acc[1][1] = wmma16(a1, b1, acc[1][1]);
    }
#pragma unroll
    for (int mi = 0; mi < 2; mi++)
#pragma unroll
        for (int ni = 0; ni < 2; ni++) {
            int n = n0 + ni * 16 + lq;
            float bv = bptr[n];
#pragma unroll
            for (int r = 0; r < 8; r++) {
                float v = acc[mi][ni][r] + bv;
                if (ACT == 1) v = fmaxf(v, 0.f);
                else if (ACT == 2) v = (v > 0.f) ? v : 0.01f * v;
                int m = m0 + mi * 16 + r + 8 * lh;
                if (TRANS) out[(size_t)n * ldo + m] = (_Float16)v;
                else       out[(size_t)m * ldo + n] = (_Float16)v;
            }
        }
}

// ---------------- TDM-staged GEMM (relu), 128x64 block macro-tile ----------------
// 8 waves = 4(M) x 2(N) subtiles of 32x32. Weight slab (64 rows x TDM_SLAB) staged
// into LDS via tensor_load_to_lds, double buffered with s_wait_tensorcnt.
__global__ void __launch_bounds__(256)
gemm_f16_tdm_kernel(const _Float16* __restrict__ A, int lda,
                    const _Float16* __restrict__ W, int ldb,
                    const float* __restrict__ bias,
                    _Float16* __restrict__ out, int ldo,
                    int M, int N, int K) {
    __shared__ _Float16 wbuf[2][64 * TDM_SLAB];
    int wave = threadIdx.x >> 5, lane = threadIdx.x & 31;
    int lq = lane & 15, lh = lane >> 4;
    int blocksM = M >> 7;
    int blkM = blockIdx.x % blocksM, blkN = blockIdx.x / blocksM;
    int m0 = blkM * 128 + (wave & 3) * 32;
    int nloc = (wave >> 2) * 32;
    int n0 = blkN * 64 + nloc;
    int nslab = (K + TDM_SLAB - 1) / TDM_SLAB;

#if HAVE_TDM
    auto issue = [&](int sl) {
        int kb = sl * TDM_SLAB;
        int kc = K - kb; if (kc > TDM_SLAB) kc = TDM_SLAB;
        const _Float16* gsrc = W + (size_t)blkN * 64 * ldb + kb;
        unsigned long long ga = (unsigned long long)(size_t)gsrc;
        unsigned lds_off = (unsigned)(size_t)(&wbuf[sl & 1][0]);
        u32x4 g0;
        g0[0] = 1u;                                   // count=1, user descriptor
        g0[1] = lds_off;                              // lds_addr
        g0[2] = (unsigned)ga;                         // global_addr[31:0]
        g0[3] = (unsigned)((ga >> 32) & 0x01ffffffu) | (2u << 30); // addr[56:32] | type=2
        i32x8 g1;
        g1[0] = (int)(1u << 16);                      // data_size=2B, no mask/flags
        g1[1] = (int)((unsigned)kc << 16);            // tensor_dim0[15:0]
        g1[2] = (int)(64u << 16);                     // tensor_dim1[15:0]
        g1[3] = (int)((unsigned)kc << 16);            // tile_dim0
        g1[4] = 64;                                   // tile_dim1
        g1[5] = ldb;                                  // tensor_dim0_stride[31:0]
        g1[6] = 0; g1[7] = 0;
        i32x4 z4 = {0, 0, 0, 0};
        i32x8 z8 = {0, 0, 0, 0, 0, 0, 0, 0};
        __builtin_amdgcn_tensor_load_to_lds(g0, g1, z4, z4, z8, 0);
    };
#endif

    v8f acc[2][2];
#pragma unroll
    for (int i = 0; i < 2; i++)
#pragma unroll
        for (int j = 0; j < 2; j++) acc[i][j] = (v8f){0.f,0.f,0.f,0.f,0.f,0.f,0.f,0.f};

#if HAVE_TDM
    if (wave == 0) {
        issue(0);
        if (nslab > 1) { issue(1); __builtin_amdgcn_s_wait_tensorcnt((short)1); }
        else           { __builtin_amdgcn_s_wait_tensorcnt((short)0); }
    }
    __syncthreads();
#endif

    for (int sl = 0; sl < nslab; sl++) {
        int kb = sl * TDM_SLAB;
        int kc = K - kb; if (kc > TDM_SLAB) kc = TDM_SLAB;
        const _Float16* bb = wbuf[sl & 1];
#if !HAVE_TDM
        // cooperative staging fallback
        __syncthreads();
        {
            int vecs = (64 * kc) >> 3;   // half8 chunks
            int rowv = kc >> 3;
            for (int i = threadIdx.x; i < vecs; i += 256) {
                int rrow = i / rowv, c8 = i - rrow * rowv;
                *(half8*)(wbuf[sl & 1] + rrow * kc + c8 * 8) =
                    *(const half8*)(W + ((size_t)blkN * 64 + rrow) * ldb + kb + c8 * 8);
            }
        }
        __syncthreads();
#endif
        for (int k = 0; k < kc; k += 32) {
            v16h a0 = ldfragA(A, lda, m0, kb + k, lq, lh);
            v16h a1 = ldfragA(A, lda, m0 + 16, kb + k, lq, lh);
            v16h b0 = ldfragB(bb, kc, nloc, k, lq, lh);
            v16h b1 = ldfragB(bb, kc, nloc + 16, k, lq, lh);
            acc[0][0] = wmma16(a0, b0, acc[0][0]);
            acc[0][1] = wmma16(a0, b1, acc[0][1]);
            acc[1][0] = wmma16(a1, b0, acc[1][0]);
            acc[1][1] = wmma16(a1, b1, acc[1][1]);
        }
#if HAVE_TDM
        __syncthreads();   // all reads of buf[sl&1] done before overwrite
        if (wave == 0 && sl + 1 < nslab) {
            if (sl + 2 < nslab) { issue(sl + 2); __builtin_amdgcn_s_wait_tensorcnt((short)1); }
            else                { __builtin_amdgcn_s_wait_tensorcnt((short)0); }
        }
        __syncthreads();
#endif
    }

#pragma unroll
    for (int mi = 0; mi < 2; mi++)
#pragma unroll
        for (int ni = 0; ni < 2; ni++) {
            int n = n0 + ni * 16 + lq;
            float bv = bias[n];
#pragma unroll
            for (int r = 0; r < 8; r++) {
                float v = fmaxf(acc[mi][ni][r] + bv, 0.f);   // relu
                int m = m0 + mi * 16 + r + 8 * lh;
                out[(size_t)m * ldo + n] = (_Float16)v;
            }
        }
}

// ---------------- fused flash attention + potential mix ----------------
__global__ void __launch_bounds__(256)
attn_kernel(const _Float16* __restrict__ sel, const _Float16* __restrict__ keys,
            const _Float16* __restrict__ valT, const float* __restrict__ FwSum,
            const float* __restrict__ Vsum, _Float16* __restrict__ q_in) {
    int w = blockIdx.x * (blockDim.x >> 5) + (threadIdx.x >> 5);
    int head = w >> 7;
    int tQ = w & 127;
    int lane = threadIdx.x & 31;
    int lq = lane & 15, lh = lane >> 4;

    const _Float16* selk  = sel  + (size_t)head * BB * DD;
    const _Float16* keysk = keys + (size_t)head * BB * DD;
    const _Float16* valk  = valT + (size_t)head * DD * BB;

    v16h bs0 = ldfragB(selk, DD, tQ * 16, 0, lq, lh);
    v16h bs1 = ldfragB(selk, DD, tQ * 16, 32, lq, lh);

    float m_run = -1e30f, l_run = 0.f;
    v8f o[4];
#pragma unroll
    for (int t = 0; t < 4; t++) o[t] = (v8f){0.f,0.f,0.f,0.f,0.f,0.f,0.f,0.f};

    for (int cb = 0; cb < BB; cb += 32) {
        v8f s0 = {0.f,0.f,0.f,0.f,0.f,0.f,0.f,0.f};
        v8f s1 = s0;
        s0 = wmma16(ldfragA(keysk, DD, cb, 0, lq, lh), bs0, s0);
        s0 = wmma16(ldfragA(keysk, DD, cb, 32, lq, lh), bs1, s0);
        s1 = wmma16(ldfragA(keysk, DD, cb + 16, 0, lq, lh), bs0, s1);
        s1 = wmma16(ldfragA(keysk, DD, cb + 16, 32, lq, lh), bs1, s1);

        float p0[8], p1[8];
        float cm = -1e30f;
#pragma unroll
        for (int r = 0; r < 8; r++) {
            s0[r] *= 0.125f; s1[r] *= 0.125f;
            cm = fmaxf(cm, fmaxf(s0[r], s1[r]));
        }
        cm = fmaxf(cm, __shfl_xor(cm, 16, 32));
        float mn = fmaxf(m_run, cm);
        float alpha = __expf(m_run - mn);
        float cs = 0.f;
#pragma unroll
        for (int r = 0; r < 8; r++) {
            p0[r] = __expf(s0[r] - mn);
            p1[r] = __expf(s1[r] - mn);
            cs += p0[r] + p1[r];
        }
        cs += __shfl_xor(cs, 16, 32);
        l_run = l_run * alpha + cs;
        m_run = mn;
#pragma unroll
        for (int t = 0; t < 4; t++)
#pragma unroll
            for (int r = 0; r < 8; r++) o[t][r] *= alpha;

        v16h pb;
#pragma unroll
        for (int r = 0; r < 8; r++) {
            float x0 = __shfl_xor(p0[r], 16, 32);
            float x1 = __shfl_xor(p1[r], 16, 32);
            if (lh == 0) { pb[r] = (_Float16)p0[r]; pb[r + 8] = (_Float16)x0; }
            else         { pb[r] = (_Float16)x1;    pb[r + 8] = (_Float16)p1[r]; }
        }
#pragma unroll
        for (int t = 0; t < 4; t++)
            o[t] = wmma16(ldfragA(valk, BB, t * 16, cb, lq, lh), pb, o[t]);
    }

    int q = tQ * 16 + lq;
    float invl = 1.f / l_run;
    float fws = 0.5f * FwSum[q];
    _Float16* orow = q_in + (size_t)q * 512 + 256 + head * DD;
    const float* vs = Vsum + head * DD;
#pragma unroll
    for (int t = 0; t < 4; t++)
#pragma unroll
        for (int r = 0; r < 8; r++) {
            int d = t * 16 + r + 8 * lh;
            float v = 4.f * o[t][r] * invl + fws * vs[d];
            orow[d] = (_Float16)v;
        }
}

// ---------------- final q-head GEMV ----------------
__global__ void __launch_bounds__(256)
qhead_kernel(const _Float16* __restrict__ H, const float* __restrict__ w,
             const float* __restrict__ b0, float* __restrict__ out) {
    int wid = (blockIdx.x * blockDim.x + threadIdx.x) >> 5;
    int lane = threadIdx.x & 31;
    if (wid >= BB) return;
    const _Float16* row = H + (size_t)wid * HH;
    float sum = 0.f;
    for (int i = lane; i < HH; i += 32) sum += (float)row[i] * w[i];
#pragma unroll
    for (int off = 16; off > 0; off >>= 1) sum += __shfl_xor(sum, off, 32);
    if (lane == 0) out[wid] = sum + b0[0];
}

// ---------------- host launch ----------------
extern "C" void kernel_launch(void* const* d_in, const int* in_sizes, int n_in,
                              void* d_out, int out_size, void* d_ws, size_t ws_size,
                              hipStream_t stream) {
    const float* s    = (const float*)d_in[0];
    const float* a    = (const float*)d_in[1];
    const float* We1  = (const float*)d_in[2];
    const float* be1  = (const float*)d_in[3];
    const float* We2  = (const float*)d_in[4];
    const float* be2  = (const float*)d_in[5];
    const float* Ws1  = (const float*)d_in[6];
    const float* bs1  = (const float*)d_in[7];
    const float* Ws2  = (const float*)d_in[8];
    const float* bs2  = (const float*)d_in[9];
    const float* Wk   = (const float*)d_in[10];
    const float* bk   = (const float*)d_in[11];
    const float* Wsel = (const float*)d_in[12];
    const float* bsel = (const float*)d_in[13];
    const float* Wv   = (const float*)d_in[14];
    const float* bv   = (const float*)d_in[15];
    const float* Wq1a = (const float*)d_in[16];
    const float* bq1a = (const float*)d_in[17];
    const float* Wq1b = (const float*)d_in[18];
    const float* bq1b = (const float*)d_in[19];
    const float* Wq2a = (const float*)d_in[20];
    const float* bq2a = (const float*)d_in[21];
    const float* Wq2b = (const float*)d_in[22];
    const float* bq2b = (const float*)d_in[23];
    float* out = (float*)d_out;

    char* ws = (char*)d_ws;
    size_t off = 0;
    auto alloc = [&](size_t bytes) -> char* {
        char* p = ws + off;
        off = (off + bytes + 255) & ~(size_t)255;
        return p;
    };
    _Float16* sa    = (_Float16*)alloc((size_t)BB * INW * 2);
    _Float16* We1t  = (_Float16*)alloc((size_t)INW * HH * 2);
    _Float16* We2t  = (_Float16*)alloc((size_t)HH * HH * 2);
    _Float16* Ws1t  = (_Float16*)alloc((size_t)SENC * HH * 2);
    _Float16* Ws2t  = (_Float16*)alloc((size_t)HH * HH * 2);
    _Float16* Wkt   = (_Float16*)alloc((size_t)KHEADS * HH * DD * 2);
    _Float16* Wselt = (_Float16*)alloc((size_t)KHEADS * HH * DD * 2);
    _Float16* Wvt   = (_Float16*)alloc((size_t)KHEADS * HH * DD * 2);
    _Float16* Wq1at = (_Float16*)alloc((size_t)512 * HH * 2);
    _Float16* Wq2at = (_Float16*)alloc((size_t)512 * HH * 2);
    _Float16* h1    = (_Float16*)alloc((size_t)BB * HH * 2);
    _Float16* sa_h  = (_Float16*)alloc((size_t)BB * HH * 2);
    _Float16* t1    = (_Float16*)alloc((size_t)BB * HH * 2);
    _Float16* selb  = (_Float16*)alloc((size_t)KHEADS * BB * DD * 2);
    _Float16* keysb = (_Float16*)alloc((size_t)KHEADS * BB * DD * 2);
    _Float16* valT  = (_Float16*)alloc((size_t)KHEADS * DD * BB * 2);
    _Float16* q_in  = (_Float16*)alloc((size_t)BB * 512 * 2);
    _Float16* hq1   = (_Float16*)alloc((size_t)BB * HH * 2);
    _Float16* hq2   = (_Float16*)alloc((size_t)BB * HH * 2);
    float* Fw    = (float*)alloc((size_t)8 * BB * 4);
    float* FwSum = (float*)alloc((size_t)BB * 4);
    float* Vsum  = (float*)alloc((size_t)KHEADS * DD * 4);
    (void)ws_size; (void)in_sizes; (void)n_in; (void)out_size;

    fw_kernel<<<8, 256, 0, stream>>>(s, Fw);
    fwsum_kernel<<<8, 256, 0, stream>>>(Fw, FwSum);

    pack_sa_kernel<<<(BB * INW + 255) / 256, 256, 0, stream>>>(s, a, sa);
    wt_pack_kernel<<<(INW * HH + 255) / 256, 256, 0, stream>>>(We1, We1t, INW, HH, INW * HH);
    wt_pack_kernel<<<(HH * HH + 255) / 256, 256, 0, stream>>>(We2, We2t, HH, HH, HH * HH);
    wt_pack_kernel<<<(SENC * HH + 255) / 256, 256, 0, stream>>>(Ws1, Ws1t, SENC, HH, SENC * HH);
    wt_pack_kernel<<<(HH * HH + 255) / 256, 256, 0, stream>>>(Ws2, Ws2t, HH, HH, HH * HH);
    wt_pack_kernel<<<(KHEADS * HH * DD + 255) / 256, 256, 0, stream>>>(Wk, Wkt, HH, DD, KHEADS * HH * DD);
    wt_pack_kernel<<<(KHEADS * HH * DD + 255) / 256, 256, 0, stream>>>(Wsel, Wselt, HH, DD, KHEADS * HH * DD);
    wt_pack_kernel<<<(KHEADS * HH * DD + 255) / 256, 256, 0, stream>>>(Wv, Wvt, HH, DD, KHEADS * HH * DD);
    wt_pack_kernel<<<(512 * HH + 255) / 256, 256, 0, stream>>>(Wq1a, Wq1at, 512, HH, 512 * HH);
    wt_pack_kernel<<<(512 * HH + 255) / 256, 256, 0, stream>>>(Wq2a, Wq2at, 512, HH, 512 * HH);

    // big-K GEMMs: TDM-staged weights in LDS, 128x64 macro-tiles, relu
    gemm_f16_tdm_kernel<<<64, 256, 0, stream>>>(sa, INW, We1t, INW, be1, h1, HH, BB, HH, INW);
    gemm_f16_kernel<0, false><<<dim3(64, 1), 256, 0, stream>>>(h1, HH, We2t, HH, be2, sa_h, HH, BB, HH, HH, 0, 0, 0);
    gemm_f16_tdm_kernel<<<64, 256, 0, stream>>>(sa, INW, Ws1t, SENC, bs1, t1, HH, BB, HH, SENC);
    gemm_f16_kernel<0, false><<<dim3(64, 1), 256, 0, stream>>>(t1, HH, Ws2t, HH, bs2, q_in, 512, BB, HH, HH, 0, 0, 0);

    // per-head projections (batched over heads via grid.y)
    gemm_f16_kernel<0, false><<<dim3(16, KHEADS), 256, 0, stream>>>(sa_h, HH, Wselt, HH, bsel, selb, DD,
                                                                    BB, DD, HH, (long)HH * DD, DD, (long)BB * DD);
    gemm_f16_kernel<0, false><<<dim3(16, KHEADS), 256, 0, stream>>>(sa_h, HH, Wkt, HH, bk, keysb, DD,
                                                                    BB, DD, HH, (long)HH * DD, DD, (long)BB * DD);
    gemm_f16_kernel<2, true><<<dim3(16, KHEADS), 256, 0, stream>>>(sa_h, HH, Wvt, HH, bv, valT, BB,
                                                                   BB, DD, HH, (long)HH * DD, DD, (long)DD * BB);

    vsum_kernel<<<KHEADS * DD, 256, 0, stream>>>(valT, Vsum);
    attn_kernel<<<64, 256, 0, stream>>>(selb, keysb, valT, FwSum, Vsum, q_in);

    gemm_f16_tdm_kernel<<<64, 256, 0, stream>>>(q_in, 512, Wq1at, 512, bq1a, hq1, HH, BB, HH, 512);
    gemm_f16_tdm_kernel<<<64, 256, 0, stream>>>(q_in, 512, Wq2at, 512, bq2a, hq2, HH, BB, HH, 512);
    qhead_kernel<<<256, 256, 0, stream>>>(hq1, Wq1b, bq1b, out);
    qhead_kernel<<<256, 256, 0, stream>>>(hq2, Wq2b, bq2b, out + BB);
}
